// GraphSAGEGraphLevel_55980603736137
// MI455X (gfx1250) — matrix operand order repaired and checked by reading, not compiled, not run on hardware
//
#include <hip/hip_runtime.h>

#define N_NODES 50000
#define N_EDGES 600000
#define N_GRAPHS 512
#define LN_EPS 1e-5f

typedef __attribute__((ext_vector_type(16))) __bf16 v16bf;
typedef __attribute__((ext_vector_type(8)))  float v8f;

__device__ __forceinline__ unsigned short f2bf(float f) {
  unsigned int u = __float_as_uint(f);
  u += 0x7FFFu + ((u >> 16) & 1u);          // round-to-nearest-even
  return (unsigned short)(u >> 16);
}

// ---------------- generic zero ----------------
__global__ void k_zero(float* __restrict__ p, long long n) {
  long long t = (long long)blockIdx.x * blockDim.x + threadIdx.x;
  if (t < n) p[t] = 0.f;
}

// ---------------- weight prepack: bf16, B-fragment order ----------------
// layout index: ((s*128 + c)*2 + g)*16 + h  <=>  W[k = s*32 + g*16 + h][c]
__global__ void k_prep_w(const float* __restrict__ Wl1, const float* __restrict__ Wr1,
                         const float* __restrict__ Wl2, const float* __restrict__ Wr2,
                         unsigned short* __restrict__ W1p, unsigned short* __restrict__ W2p) {
  int t = blockIdx.x * blockDim.x + threadIdx.x;
  if (t < 3 * 4096) {                    // layer 1: Kpad=96 -> 3 ksteps
    int h = t & 15, g = (t >> 4) & 1, c = (t >> 5) & 127, s = t >> 12;
    int k = s * 32 + g * 16 + h;
    float v = 0.f;
    if (k < 44)       v = Wl1[c * 44 + k];
    else if (k < 88)  v = Wr1[c * 44 + (k - 44)];
    W1p[t] = f2bf(v);
  } else if (t < 3 * 4096 + 8 * 4096) {  // layer 2: Kpad=256 -> 8 ksteps
    int i = t - 3 * 4096;
    int h = i & 15, g = (i >> 4) & 1, c = (i >> 5) & 127, s = i >> 12;
    int k = s * 32 + g * 16 + h;
    float v = (k < 128) ? Wl2[c * 128 + k] : Wr2[c * 128 + (k - 128)];
    W2p[i] = f2bf(v);
  }
}

// ---------------- node feature assembly: h0 = [x | xdims | st_emb[xsttype]] ----------------
__global__ void k_build_h0(const float* __restrict__ x, const float* __restrict__ xd,
                           const int* __restrict__ st, const float* __restrict__ emb,
                           float* __restrict__ h0) {
  long long t = (long long)blockIdx.x * blockDim.x + threadIdx.x;
  if (t >= (long long)N_NODES * 44) return;
  int n = (int)(t / 44), d = (int)(t % 44);
  float v;
  if (d < 29)      v = x[n * 29 + d];
  else if (d < 32) v = xd[n * 3 + (d - 29)];
  else             v = emb[st[n] * 12 + (d - 32)];
  h0[t] = v;
}

// ---------------- edge scatter (float4): agg[dst] += h[src]; cnt[dst] += 1 ----------------
// Dq = D/4; rows of both 44-float (176B) and 128-float (512B) are 16B multiples.
__global__ void k_scatter4(const float* __restrict__ h, const int* __restrict__ src,
                           const int* __restrict__ dst, float* __restrict__ agg,
                           float* __restrict__ cnt, int Dq) {
  long long t = (long long)blockIdx.x * blockDim.x + threadIdx.x;
  if (t >= (long long)N_EDGES * Dq) return;
  int e = (int)(t / Dq), q = (int)(t - (long long)e * Dq);
  int s = src[e], dd = dst[e];
  const float4 v = *(const float4*)(h + ((long long)s * Dq + q) * 4);
  float* a = agg + ((long long)dd * Dq + q) * 4;
  atomicAdd(a + 0, v.x);
  atomicAdd(a + 1, v.y);
  atomicAdd(a + 2, v.z);
  atomicAdd(a + 3, v.w);
  if (q == 0 && cnt) atomicAdd(&cnt[dd], 1.0f);
}

// ---------------- A assembly: bf16 [mean | h | pad] ----------------
__global__ void k_build_A(const float* __restrict__ agg, const float* __restrict__ cnt,
                          const float* __restrict__ hprev, unsigned short* __restrict__ A,
                          int Dm, int Dh, int Kpad) {
  long long t = (long long)blockIdx.x * blockDim.x + threadIdx.x;
  if (t >= (long long)N_NODES * Kpad) return;
  int n = (int)(t / Kpad), k = (int)(t % Kpad);
  float v = 0.f;
  if (k < Dm)            v = agg[(long long)n * Dm + k] / fmaxf(cnt[n], 1.0f);
  else if (k < Dm + Dh)  v = hprev[(long long)n * Dh + (k - Dm)];
  A[t] = f2bf(v);
}

// ---------------- fused WMMA GEMM + bias + LayerNorm + ReLU ----------------
// block = 256 threads = 8 waves; block computes 16 rows x 128 cols.
// A tile (16 x Kpad bf16) staged in LDS ONCE per block (8x traffic cut vs per-wave
// global loads); wave w owns 16x16 tile at cols [16w, 16w+16); K loop of
// v_wmma_f32_16x16x32_bf16 with A fragments from ds_load_b128.
__global__ __launch_bounds__(256) void k_gemm_ln_relu(
    const unsigned short* __restrict__ A,   // [M][Kpad] bf16
    const unsigned short* __restrict__ Wf,  // fragment-packed [ksteps][128][2][16] bf16
    const float* __restrict__ bias, const float* __restrict__ gamma,
    const float* __restrict__ beta, float* __restrict__ out,  // [M][128]
    int Kpad, int ksteps) {
  __shared__ __align__(16) unsigned short ldsA[16 * 256];  // row stride fixed at 256 halfs
  __shared__ float tile[16][128];
  __shared__ float smu[16], srs[16];

  int wave = threadIdx.x >> 5, lane = threadIdx.x & 31;
  int rl = lane & 15, kh = lane >> 4;        // row-in-tile / K-half select
  long long row0 = (long long)blockIdx.x * 16;
  int col0 = wave * 16;

  // cooperative A-tile stage: global b128 -> LDS b128
  int qpr = Kpad >> 3;                        // uint4 quads per row
  for (int idx = threadIdx.x; idx < 16 * qpr; idx += 256) {
    int row = idx / qpr, q = idx - row * qpr;
    *(uint4*)(&ldsA[row * 256 + q * 8]) =
        *(const uint4*)(A + (row0 + row) * Kpad + q * 8);
  }
  __syncthreads();

  v8f acc = {0.f, 0.f, 0.f, 0.f, 0.f, 0.f, 0.f, 0.f};
  const unsigned short* arow = &ldsA[rl * 256];

  for (int s = 0; s < ksteps; ++s) {
    int kb = s * 32;
    union { v16bf v; uint4 q[2]; } fa, fb;
    // A 16x32 bf16 fragment from LDS (lanes0-15 K[kb..kb+15], lanes16-31 K[kb+16..kb+31])
    fa.q[0] = *(const uint4*)(arow + kb + 16 * kh);
    fa.q[1] = *(const uint4*)(arow + kb + 16 * kh + 8);
    // B 32x16 bf16 fragment, contiguous thanks to prepack (L2-resident, <=64KB total)
    const unsigned short* bp = Wf + (((s * 128 + col0 + rl) * 2 + kh) << 4);
    fb.q[0] = *(const uint4*)(bp);
    fb.q[1] = *(const uint4*)(bp + 8);
    acc = __builtin_amdgcn_wmma_f32_16x16x32_bf16(false, fa.v, false, fb.v,
                                                  (short)0, acc, false, false);
  }

  // C layout: VGPR r -> M = r + 8*kh, N = col0 + rl. Add bias into LDS tile.
#pragma unroll
  for (int r = 0; r < 8; ++r) {
    int m = r + 8 * kh, nn = col0 + rl;
    tile[m][nn] = acc[r] + bias[nn];
  }
  __syncthreads();

  if (threadIdx.x < 16) {
    float s1 = 0.f, s2 = 0.f;
    for (int c = 0; c < 128; ++c) { float h = tile[threadIdx.x][c]; s1 += h; s2 += h * h; }
    float mu = s1 * (1.f / 128.f);
    float var = s2 * (1.f / 128.f) - mu * mu;
    smu[threadIdx.x] = mu;
    srs[threadIdx.x] = rsqrtf(var + LN_EPS);
  }
  __syncthreads();

#pragma unroll
  for (int i = 0; i < 8; ++i) {
    int idx = i * 256 + threadIdx.x;       // coalesced stores
    int row = idx >> 7, c = idx & 127;
    float h = tile[row][c];
    float y = gamma[c] * (h - smu[row]) * srs[row] + beta[c];
    out[(row0 + row) * 128 + c] = fmaxf(y, 0.f);
  }
}

// ---------------- graph pooling: sum / count / max (post-ReLU => uint-bit max valid) ---------
__global__ void k_pool(const float* __restrict__ h2, const int* __restrict__ batch,
                       float* __restrict__ hs, unsigned int* __restrict__ hmax,
                       float* __restrict__ nc) {
  long long t = (long long)blockIdx.x * blockDim.x + threadIdx.x;
  if (t >= (long long)N_NODES * 128) return;
  int n = (int)(t >> 7), d = (int)(t & 127);
  float v = h2[t];
  int g = batch[n];
  atomicAdd(&hs[g * 128 + d], v);
  atomicMax(&hmax[g * 128 + d], __float_as_uint(v));
  if (d == 0) atomicAdd(&nc[g], 1.0f);
}

// ---------------- final tiny GEMM: pooled[512,256] @ Wlin^T + blin ----------------
__global__ void k_final(const float* __restrict__ hs, const unsigned int* __restrict__ hmax,
                        const float* __restrict__ nc, const float* __restrict__ Wlin,
                        const float* __restrict__ blin, float* __restrict__ out) {
  int t = blockIdx.x * blockDim.x + threadIdx.x;
  if (t >= N_GRAPHS * 10) return;
  int g = t / 10, o = t % 10;
  float inv = 1.f / fmaxf(nc[g], 1.f);
  float acc = blin[o];
  for (int d = 0; d < 128; ++d) acc += hs[g * 128 + d] * inv * Wlin[o * 256 + d];
  for (int d = 0; d < 128; ++d) acc += __uint_as_float(hmax[g * 128 + d]) * Wlin[o * 256 + 128 + d];
  out[t] = acc;
}

extern "C" void kernel_launch(void* const* d_in, const int* in_sizes, int n_in,
                              void* d_out, int out_size, void* d_ws, size_t ws_size,
                              hipStream_t stream) {
  (void)in_sizes; (void)n_in; (void)out_size; (void)ws_size;
  const float* x      = (const float*)d_in[0];
  const float* xdims  = (const float*)d_in[1];
  const int*   sttype = (const int*)d_in[2];
  const int*   ei     = (const int*)d_in[3];
  const int*   batch  = (const int*)d_in[4];
  const float* st_emb = (const float*)d_in[5];
  const float* Wl1 = (const float*)d_in[6],  *bl1 = (const float*)d_in[7];
  const float* Wr1 = (const float*)d_in[8],  *g1  = (const float*)d_in[9],  *be1 = (const float*)d_in[10];
  const float* Wl2 = (const float*)d_in[11], *bl2 = (const float*)d_in[12];
  const float* Wr2 = (const float*)d_in[13], *g2  = (const float*)d_in[14], *be2 = (const float*)d_in[15];
  const float* Wlin = (const float*)d_in[16], *blin = (const float*)d_in[17];
  float* out = (float*)d_out;

  // workspace layout
  char* w = (char*)d_ws;
  size_t off = 0;
  auto alloc = [&](size_t bytes) -> char* {
    char* p = w + off;
    off = (off + bytes + 255) & ~(size_t)255;
    return p;
  };
  float*          cnt  = (float*)alloc((size_t)N_NODES * 4);
  float*          h0   = (float*)alloc((size_t)N_NODES * 44 * 4);
  float*          agg  = (float*)alloc((size_t)N_NODES * 128 * 4);
  unsigned short* Abuf = (unsigned short*)alloc((size_t)N_NODES * 256 * 2);
  float*          h1   = (float*)alloc((size_t)N_NODES * 128 * 4);
  unsigned short* W1p  = (unsigned short*)alloc((size_t)3 * 4096 * 2);
  unsigned short* W2p  = (unsigned short*)alloc((size_t)8 * 4096 * 2);
  float*          hs   = (float*)alloc((size_t)N_GRAPHS * 128 * 4);
  unsigned int*   hmax = (unsigned int*)alloc((size_t)N_GRAPHS * 128 * 4);
  float*          nc   = (float*)alloc((size_t)N_GRAPHS * 4);
  float*          h2   = agg;  // agg is dead after A2 assembly; reuse for h2

  const int* src = ei;
  const int* dst = ei + N_EDGES;

  auto G = [](long long n, int b) { return (unsigned)((n + b - 1) / b); };

  // weights -> bf16 fragment order
  k_prep_w<<<G(11 * 4096, 256), 256, 0, stream>>>(Wl1, Wr1, Wl2, Wr2, W1p, W2p);
  // h0 assembly
  k_build_h0<<<G((long long)N_NODES * 44, 256), 256, 0, stream>>>(x, xdims, sttype, st_emb, h0);

  // ---- layer 1 ----
  k_zero<<<G(N_NODES, 256), 256, 0, stream>>>(cnt, N_NODES);
  k_zero<<<G((long long)N_NODES * 44, 256), 256, 0, stream>>>(agg, (long long)N_NODES * 44);
  k_scatter4<<<G((long long)N_EDGES * 11, 256), 256, 0, stream>>>(h0, src, dst, agg, cnt, 11);
  k_build_A<<<G((long long)N_NODES * 96, 256), 256, 0, stream>>>(agg, cnt, h0, Abuf, 44, 44, 96);
  k_gemm_ln_relu<<<N_NODES / 16, 256, 0, stream>>>(Abuf, W1p, bl1, g1, be1, h1, 96, 3);

  // ---- layer 2 ----
  k_zero<<<G((long long)N_NODES * 128, 256), 256, 0, stream>>>(agg, (long long)N_NODES * 128);
  k_scatter4<<<G((long long)N_EDGES * 32, 256), 256, 0, stream>>>(h1, src, dst, agg, nullptr, 32);
  k_build_A<<<G((long long)N_NODES * 256, 256), 256, 0, stream>>>(agg, cnt, h1, Abuf, 128, 128, 256);
  k_zero<<<G((long long)N_GRAPHS * 128, 256), 256, 0, stream>>>(hs, (long long)N_GRAPHS * 128);
  k_zero<<<G((long long)N_GRAPHS * 128, 256), 256, 0, stream>>>((float*)hmax, (long long)N_GRAPHS * 128);
  k_zero<<<G(N_GRAPHS, 256), 256, 0, stream>>>(nc, N_GRAPHS);
  k_gemm_ln_relu<<<N_NODES / 16, 256, 0, stream>>>(Abuf, W2p, bl2, g2, be2, h2, 256, 8);

  // ---- pooling + readout ----
  k_pool<<<G((long long)N_NODES * 128, 256), 256, 0, stream>>>(h2, batch, hs, hmax, nc);
  k_final<<<G(N_GRAPHS * 10, 256), 256, 0, stream>>>(hs, hmax, nc, Wlin, blin, out);
}
